// RealNVP_83090437308572
// MI455X (gfx1250) — matrix operand order repaired and checked by reading, not compiled, not run on hardware
//
#include <hip/hip_runtime.h>

typedef __attribute__((ext_vector_type(16))) _Float16 v16h;
typedef __attribute__((ext_vector_type(8)))  _Float16 v8h;
typedef __attribute__((ext_vector_type(2)))  _Float16 h2;
typedef __attribute__((ext_vector_type(2)))  __fp16   f16x2;
typedef __attribute__((ext_vector_type(8)))  float    v8f;
typedef __attribute__((ext_vector_type(4)))  float    v4f;

#define LATENT   64
#define NLAYER   16
#define HIDDEN   16
#define BATCHN   262144
#define NEGSLOPE 0.01f

#define WAVES_PER_BLOCK 16
#define THREADS   (WAVES_PER_BLOCK * 32)   // 512
#define BLOCKS    256
#define NTILES    (BATCHN / 16)            // 16384 row tiles
#define TILES_PER_WAVE (NTILES / (BLOCKS * WAVES_PER_BLOCK)) // 4

// padded LDS strides (bank-conflict avoidance; keep 16B vector alignment)
#define W1S 72      // halves per row of W1stage [L][16][72] (bias baked at k=q)
#define W2S 40      // halves per row of W2c     [L][32][40] (bias baked at k=16)
#define ZS  68      // floats per row of zbuf    [16][68]
#define HS  20      // floats per row of hbuf    [16][20] (col-major [k][m])

// ---- LDS layout (bytes) ----
#define OFF_W1S   0
#define OFF_W1T   (OFF_W1S + NLAYER*HIDDEN*W1S*2)   // 36864
#define OFF_W2Sc  (OFF_W1T + NLAYER*HIDDEN*W1S*2)   // 73728
#define OFF_W2Tc  (OFF_W2Sc + NLAYER*32*W2S*2)      // 114688
#define OFF_SCR   (OFF_W2Tc + NLAYER*32*W2S*2)      // 155648
// per-wave scratch: zbuf 16*68*4=4352 + hS 1280 + hT 1280 = 6912 B
#define SCR_PER_WAVE 6912
#define SMEM_BYTES (OFF_SCR + WAVES_PER_BLOCK*SCR_PER_WAVE) // 266240 (< 320KB)

__device__ __forceinline__ void lds_wave_fence() {
    __asm__ volatile("s_wait_dscnt 0" ::: "memory");
}

__device__ __forceinline__ float fast_tanh(float x) {
#if __has_builtin(__builtin_amdgcn_tanhf)
    return __builtin_amdgcn_tanhf(x);      // gfx1250 v_tanh_f32 (TRANS unit)
#else
    const float e = __expf(2.0f * x);
    return 1.0f - 2.0f * __builtin_amdgcn_rcpf(e + 1.0f);
#endif
}

// one v_cvt_pk_rtz_f16_f32 per f16 pair
__device__ __forceinline__ h2 cvt_pk(float a, float b) {
#if __has_builtin(__builtin_amdgcn_cvt_pkrtz)
    const f16x2 t = __builtin_amdgcn_cvt_pkrtz(a, b);
    return __builtin_bit_cast(h2, t);
#else
    return h2{(_Float16)a, (_Float16)b};
#endif
}

__device__ __forceinline__ v8f wmma_f16(v16h a, v16h b, v8f c) {
    return __builtin_amdgcn_wmma_f32_16x16x32_f16(false, a, false, b, (short)0, c,
                                                  false, false);
}

// A fragment of masked z (16x32 f16), K-half c, parity P compile-time.
// Dead-parity K rows of W1stage are zero, so dead A elements are don't-care;
// only the bias column k=q (pair 0, dead half; aliases hit staged-zero rows)
// must be exactly 1.0. One v_cvt_pk_rtz per pair.
template <int P>
__device__ __forceinline__ v16h load_a_z(const float* zrow, int c, int hh) {
    const float* base = zrow + c*32 + 16*hh;
    const v4f f0 = *(const v4f*)(base + 0);
    const v4f f1 = *(const v4f*)(base + 4);
    const v4f f2 = *(const v4f*)(base + 8);
    const v4f f3 = *(const v4f*)(base + 12);
    union { v16h v; h2 h[8]; } u;
    if (P == 0) {   // live = even K = low half of each pair; bias col k=1
        u.h[0] = cvt_pk(f0[0], 1.f);
        u.h[1] = cvt_pk(f0[2], 0.f);
        u.h[2] = cvt_pk(f1[0], 0.f);
        u.h[3] = cvt_pk(f1[2], 0.f);
        u.h[4] = cvt_pk(f2[0], 0.f);
        u.h[5] = cvt_pk(f2[2], 0.f);
        u.h[6] = cvt_pk(f3[0], 0.f);
        u.h[7] = cvt_pk(f3[2], 0.f);
    } else {        // live = odd K = high half; bias col k=0
        u.h[0] = cvt_pk(1.f, f0[1]);
        u.h[1] = cvt_pk(0.f, f0[3]);
        u.h[2] = cvt_pk(0.f, f1[1]);
        u.h[3] = cvt_pk(0.f, f1[3]);
        u.h[4] = cvt_pk(0.f, f2[1]);
        u.h[5] = cvt_pk(0.f, f2[3]);
        u.h[6] = cvt_pk(0.f, f3[1]);
        u.h[7] = cvt_pk(0.f, f3[3]);
    }
    return u.v;
}

// A fragment for GEMM2 from col-major h buffer [k][m]; element e=8 (k=16) is
// the bias-1 column (hh=1 alias k=24 hits staged-zero rows), rest of the
// K>=16 pad is compile-time zero.
__device__ __forceinline__ v16h load_a_h(const float* hb, int n, int hh) {
    union { v16h v; h2 h[8]; } u;
#pragma unroll
    for (int i = 0; i < 4; ++i) {
        const float a = hb[(8*hh + 2*i    )*HS + n];
        const float b = hb[(8*hh + 2*i + 1)*HS + n];
        u.h[i] = cvt_pk(a, b);
    }
    u.h[4] = h2{(_Float16)1.f, (_Float16)0.f};
#pragma unroll
    for (int i = 5; i < 8; ++i) u.h[i] = h2{(_Float16)0.f, (_Float16)0.f};
    return u.v;
}

template <int P>
__device__ __forceinline__ void layer_step(int lay,
        const _Float16* __restrict__ wS1, const _Float16* __restrict__ wT1,
        const _Float16* __restrict__ wS2, const _Float16* __restrict__ wT2,
        float* __restrict__ zbuf, float* __restrict__ hSb, float* __restrict__ hTb,
        int n, int hh, float* __restrict__ ldacc) {
    lds_wave_fence();                       // prior z stores visible

    // ---- A fragments of masked z (bias column baked in) ----
    const float* zrow = zbuf + n*ZS;
    const v16h aZ0 = load_a_z<P>(zrow, 0, hh);
    const v16h aZ1 = load_a_z<P>(zrow, 1, hh);

    // ---- GEMM1 both nets (C starts at inline 0; bias arrives via k=q) ----
    const _Float16* w1sl = wS1 + lay*(HIDDEN*W1S) + n*W1S + 16*hh;
    const _Float16* w1tl = wT1 + lay*(HIDDEN*W1S) + n*W1S + 16*hh;
    v8f hcS{}, hcT{};
    hcS = wmma_f16(aZ0, *(const v16h*)(w1sl +  0), hcS);
    hcS = wmma_f16(aZ1, *(const v16h*)(w1sl + 32), hcS);
    hcT = wmma_f16(aZ0, *(const v16h*)(w1tl +  0), hcT);
    hcT = wmma_f16(aZ1, *(const v16h*)(w1tl + 32), hcT);

    // LeakyReLU: max(x, 0.01x) (slope < 1)
#pragma unroll
    for (int r = 0; r < 8; ++r) {
        hcS[r] = fmaxf(hcS[r], NEGSLOPE * hcS[r]);
        hcT[r] = fmaxf(hcT[r], NEGSLOPE * hcT[r]);
    }

    // ---- h -> LDS col-major [k=n][m=8hh+r] ----
    *(v4f*)&hSb[n*HS + 8*hh + 0] = v4f{hcS[0], hcS[1], hcS[2], hcS[3]};
    *(v4f*)&hSb[n*HS + 8*hh + 4] = v4f{hcS[4], hcS[5], hcS[6], hcS[7]};
    *(v4f*)&hTb[n*HS + 8*hh + 0] = v4f{hcT[0], hcT[1], hcT[2], hcT[3]};
    *(v4f*)&hTb[n*HS + 8*hh + 4] = v4f{hcT[4], hcT[5], hcT[6], hcT[7]};
    lds_wave_fence();

    const v16h ahS = load_a_h(hSb, n, hh);
    const v16h ahT = load_a_h(hTb, n, hh);

    // ---- GEMM2 on the 32 transformed columns (2 N-tiles, bias via k=16) ----
    const _Float16* w2sl = wS2 + lay*(32*W2S) + n*W2S + 8*hh;
    const _Float16* w2tl = wT2 + lay*(32*W2S) + n*W2S + 8*hh;
    v8f oS0{}, oS1{}, oT0{}, oT1{};
    oS0 = wmma_f16(ahS, __builtin_shufflevector(*(const v8h*)(w2sl + 0*16*W2S), *(const v8h*)(w2sl + 0*16*W2S + 16), 0,1,2,3,4,5,6,7,8,9,10,11,12,13,14,15), oS0);
    oS1 = wmma_f16(ahS, __builtin_shufflevector(*(const v8h*)(w2sl + 1*16*W2S), *(const v8h*)(w2sl + 1*16*W2S + 16), 0,1,2,3,4,5,6,7,8,9,10,11,12,13,14,15), oS1);
    oT0 = wmma_f16(ahT, __builtin_shufflevector(*(const v8h*)(w2tl + 0*16*W2S), *(const v8h*)(w2tl + 0*16*W2S + 16), 0,1,2,3,4,5,6,7,8,9,10,11,12,13,14,15), oT0);
    oT1 = wmma_f16(ahT, __builtin_shufflevector(*(const v8h*)(w2tl + 1*16*W2S), *(const v8h*)(w2tl + 1*16*W2S + 16), 0,1,2,3,4,5,6,7,8,9,10,11,12,13,14,15), oT1);

    // ---- tanh + affine update fused (no fragment copies); q = 1-P ----
    constexpr int q = 1 - P;
    float* zupd = zbuf + 8*hh*ZS + 2*n + q;
#pragma unroll
    for (int r = 0; r < 8; ++r) {
        const float s0 = fast_tanh(oS0[r]);
        const float s1 = fast_tanh(oS1[r]);
        zupd[r*ZS +  0] = zupd[r*ZS +  0] * __expf(s0) + oT0[r];
        zupd[r*ZS + 32] = zupd[r*ZS + 32] * __expf(s1) + oT1[r];
        ldacc[r] += s0 + s1;
    }
}

__global__ __launch_bounds__(THREADS, 1)
void realnvp_wmma_kernel(const float* __restrict__ x,
                         const float* __restrict__ gS1, const float* __restrict__ gbS1,
                         const float* __restrict__ gS2, const float* __restrict__ gbS2,
                         const float* __restrict__ gT1, const float* __restrict__ gbT1,
                         const float* __restrict__ gT2, const float* __restrict__ gbT2,
                         float* __restrict__ zout, float* __restrict__ ldout) {
    extern __shared__ __align__(16) char smem[];

    _Float16* wS1 = (_Float16*)(smem + OFF_W1S);
    _Float16* wT1 = (_Float16*)(smem + OFF_W1T);
    _Float16* wS2 = (_Float16*)(smem + OFF_W2Sc);
    _Float16* wT2 = (_Float16*)(smem + OFF_W2Tc);

    const int tid = threadIdx.x;

    // ---- stage weights once per block (f32->f16, masks + biases baked in) ----
    __builtin_prefetch(gS1, 0, 1);
    __builtin_prefetch(gT1, 0, 1);
    for (int i = tid; i < NLAYER*HIDDEN*LATENT; i += THREADS) {
        const int lay = i >> 10, rem = i & 1023;
        const int nn = rem >> 6, k = rem & 63;
        const int p = lay & 1, q = 1 - p;
        const int d = lay*(HIDDEN*W1S) + nn*W1S + k;
        float vs, vt;
        if ((k & 1) == p)      { vs = gS1[i]; vt = gT1[i]; }          // live parity
        else if (k == q)       { vs = gbS1[lay*HIDDEN + nn]; vt = gbT1[lay*HIDDEN + nn]; } // bias col
        else                   { vs = 0.f; vt = 0.f; }                // dead
        wS1[d] = (_Float16)vs;
        wT1[d] = (_Float16)vt;
    }
    // W2 compacted to transformed output parity q; k=16 row carries the bias
    for (int i = tid; i < NLAYER*32*32; i += THREADS) {
        const int lay = i >> 10, rem = i & 1023;
        const int c = rem >> 5, k = rem & 31;
        const int q = 1 - (lay & 1);
        const int col = 2*c + q;
        const int d = lay*(32*W2S) + c*W2S + k;
        float vs, vt;
        if (k < HIDDEN)       { vs = gS2[lay*LATENT*HIDDEN + col*HIDDEN + k];
                                vt = gT2[lay*LATENT*HIDDEN + col*HIDDEN + k]; }
        else if (k == HIDDEN) { vs = gbS2[lay*LATENT + col]; vt = gbT2[lay*LATENT + col]; }
        else                  { vs = 0.f; vt = 0.f; }
        wS2[d] = (_Float16)vs;
        wT2[d] = (_Float16)vt;
    }
    __syncthreads();

    const int lane = tid & 31;
    const int wave = tid >> 5;
    const int n    = lane & 15;
    const int hh   = lane >> 4;

    float* zbuf = (float*)(smem + OFF_SCR + wave*SCR_PER_WAVE);          // [16][68] f32
    float* hSb  = (float*)(smem + OFF_SCR + wave*SCR_PER_WAVE + 4352);   // [16][20]
    float* hTb  = (float*)(smem + OFF_SCR + wave*SCR_PER_WAVE + 5632);

    const int gwave = blockIdx.x * WAVES_PER_BLOCK + wave;

    for (int it = 0; it < TILES_PER_WAVE; ++it) {
        const int rowBase = (gwave * TILES_PER_WAVE + it) * 16;

        // ---- coalesced tile load: 16 rows x 64 f32 -> padded zbuf ----
#pragma unroll
        for (int j = 0; j < 8; ++j) {
            const int id = j*32 + lane;
            const int row = id >> 4, ci = id & 15;
            const v4f v = *(const v4f*)(x + (size_t)(rowBase + row)*LATENT + ci*4);
            *(v4f*)(zbuf + row*ZS + ci*4) = v;
        }

        float ldacc[8];
#pragma unroll
        for (int r = 0; r < 8; ++r) ldacc[r] = 0.f;

        for (int l2 = 0; l2 < NLAYER/2; ++l2) {
            layer_step<0>(2*l2 + 0, wS1, wT1, wS2, wT2, zbuf, hSb, hTb, n, hh, ldacc);
            layer_step<1>(2*l2 + 1, wS1, wT1, wS2, wT2, zbuf, hSb, hTb, n, hh, ldacc);
        }
        lds_wave_fence();

        // ---- coalesced tile store ----
#pragma unroll
        for (int j = 0; j < 8; ++j) {
            const int id = j*32 + lane;
            const int row = id >> 4, ci = id & 15;
            const v4f v = *(const v4f*)(zbuf + row*ZS + ci*4);
            *(v4f*)(zout + (size_t)(rowBase + row)*LATENT + ci*4) = v;
        }

        // ---- logdet: reduce across the 16 lanes of each half-wave ----
#pragma unroll
        for (int r = 0; r < 8; ++r) {
            float v = ldacc[r];
#pragma unroll
            for (int off = 1; off < 16; off <<= 1)
                v += __shfl_xor(v, off, 16);
            if (n == 0)
                ldout[rowBase + r + 8*hh] = v;
        }
    }
}

extern "C" void kernel_launch(void* const* d_in, const int* in_sizes, int n_in,
                              void* d_out, int out_size, void* d_ws, size_t ws_size,
                              hipStream_t stream) {
    const float* x    = (const float*)d_in[0];
    const float* sW1  = (const float*)d_in[1];
    const float* sb1  = (const float*)d_in[2];
    const float* sW2  = (const float*)d_in[3];
    const float* sb2  = (const float*)d_in[4];
    const float* tW1  = (const float*)d_in[5];
    const float* tb1  = (const float*)d_in[6];
    const float* tW2  = (const float*)d_in[7];
    const float* tb2  = (const float*)d_in[8];

    float* zoutp  = (float*)d_out;                     // [B,64,1] flat
    float* ldoutp = zoutp + (size_t)BATCHN * LATENT;   // [B]

    realnvp_wmma_kernel<<<dim3(BLOCKS), dim3(THREADS), SMEM_BYTES, stream>>>(
        x, sW1, sb1, sW2, sb2, tW1, tb1, tW2, tb2, zoutp, ldoutp);
}
